// DiffGraphSTFormerBlock_55405078118814
// MI455X (gfx1250) — compile-verified, matrix operand order checked by reading, not run on hardware
//
#include <hip/hip_runtime.h>
#include <cstdint>

// ---------------------------------------------------------------------------
// DiffGraphSTFormerBlock for MI455X (gfx1250): bf16 WMMA pipeline with
// async global->LDS weight DMA (GLOBAL_LOAD_ASYNC_TO_LDS_B128).
// B=8, T=128, N=64, IN=32, D=128, H=4, HD=32, TOK = B*T*N = 65536.
// ---------------------------------------------------------------------------

typedef __bf16 bf16;
typedef __attribute__((ext_vector_type(16))) __bf16 v16bf;
typedef __attribute__((ext_vector_type(8)))  __bf16 v8bf;
typedef __attribute__((ext_vector_type(8)))  float  v8f;

#if __has_builtin(__builtin_amdgcn_global_load_async_to_lds_b128)
#define HAVE_ASYNC_LDS 1
typedef int v4i_t __attribute__((vector_size(16)));
typedef __attribute__((address_space(1))) v4i_t* as1_v4i;
typedef __attribute__((address_space(3))) v4i_t* as3_v4i;
#else
#define HAVE_ASYNC_LDS 0
#endif

static __device__ __forceinline__ bf16 f2bf(float f) {
    uint32_t u = __builtin_bit_cast(uint32_t, f);
    uint32_t r = u + 0x7FFFu + ((u >> 16) & 1u);   // round-to-nearest-even
    uint16_t h = (uint16_t)(r >> 16);
    return __builtin_bit_cast(bf16, h);
}

static __device__ __forceinline__ float gelu_tanh(float x) {
    float x3 = x * x * x;
    return 0.5f * x * (1.0f + tanhf(0.7978845608028654f * (x + 0.044715f * x3)));
}

// Load a 16x32 bf16 WMMA fragment from an LDS tile with row stride LDK
// (multiple of 8).  16-bit operand layout:
//   lanes 0-15 : row = lane,    K = kb..kb+7 (VGPR0-3), kb+16..kb+23 (VGPR4-7)
//   lanes 16-31: row = lane-16, caller pre-offsets kb by +8.
template<int LDK>
static __device__ __forceinline__ v16bf load_frag(const bf16* base, int row, int kb) {
    const bf16* p = base + row * LDK + kb;
    v8bf lo = *(const v8bf*)(p);
    v8bf hi = *(const v8bf*)(p + 16);
    v16bf r;
#pragma unroll
    for (int i = 0; i < 8; ++i) { r[i] = lo[i]; r[i + 8] = hi[i]; }
    return r;
}

static __device__ __forceinline__ v8f wmma_bf16(v16bf a, v16bf b, v8f c) {
    return __builtin_amdgcn_wmma_f32_16x16x32_bf16(false, a, false, b, (short)0, c,
                                                   false, false);
}

// ---------------------------------------------------------------------------
// Weight prep: W (nodes x K x NOUT, f32) -> Wt (nodes x NOUT x (K+8), bf16),
// i.e. transposed + padded so it matches the GEMM's LDS B-tile byte-for-byte.
// ---------------------------------------------------------------------------
template<int K, int NOUT>
__global__ __launch_bounds__(256)
void prep_w(const float* __restrict__ W, bf16* __restrict__ Wt, int nodes)
{
    constexpr int LDK = K + 8;
    long idx   = (long)blockIdx.x * 256 + threadIdx.x;
    long total = (long)nodes * NOUT * LDK;
    if (idx >= total) return;
    int  t    = (int)(idx % LDK);
    long r    = idx / LDK;
    int  n    = (int)(r % NOUT);
    int  node = (int)(r / NOUT);
    Wt[idx] = (t < K) ? f2bf(W[((long)node * K + t) * NOUT + n]) : f2bf(0.f);
}

// ---------------------------------------------------------------------------
// Fused GEMM:  out = epilogue( A(rows x K) @ W(K x NOUT) + bias )
// Weights arrive pre-transposed bf16 (prep_w layout) and are DMA'd straight
// into LDS with GLOBAL_LOAD_ASYNC_TO_LDS_B128 (overlapped with the f32->bf16
// A-tile staging), falling back to vector LDS copies if the builtin is absent.
// Block: 128 threads (4 wave32); 64 output rows per block.
// ---------------------------------------------------------------------------
template<int K, int NOUT, bool PER_NODE, bool DO_GELU, bool DO_RES, bool DO_PE, bool OUT_BF>
__global__ __launch_bounds__(128)
void gemm_ep(const float* __restrict__ A, const bf16* __restrict__ Wt,
             const float* __restrict__ bias, const float* __restrict__ res,
             void* __restrict__ outv, int nodes)
{
    constexpr int LDK = K + 8;
    __shared__ __align__(16) bf16 lA[64 * LDK];
    __shared__ __align__(16) bf16 lB[NOUT * LDK];

    const int  tid  = threadIdx.x;
    const int  tile = blockIdx.x;                       // 64-row tile index
    const int  node = PER_NODE ? blockIdx.y : 0;
    const long bbase = PER_NODE ? (long)node * NOUT : 0;
    const bf16* wsrc = Wt + (PER_NODE ? (size_t)node * NOUT * LDK : 0);

    // ---- weight tile: async DMA global->LDS (LDS-identical layout) ----
#if HAVE_ASYNC_LDS
    for (int c = tid; c < NOUT * LDK / 8; c += 128)
        __builtin_amdgcn_global_load_async_to_lds_b128(
            (as1_v4i)(wsrc + c * 8), (as3_v4i)(lB + c * 8), 0, 0);
#else
    for (int c = tid; c < NOUT * LDK / 8; c += 128)
        *(v8bf*)(lB + c * 8) = *(const v8bf*)(wsrc + c * 8);
#endif

    // ---- A tile: f32 -> bf16 (overlaps the weight DMA) ----
    for (int idx = tid; idx < 64 * (K / 4); idx += 128) {
        int m  = idx / (K / 4);
        int k4 = (idx - m * (K / 4)) * 4;
        long tok = (long)tile * 64 + m;
        long row = PER_NODE ? (tok * nodes + node) : tok;
        const float4 v = *(const float4*)(A + row * K + k4);
        bf16* d = lA + m * LDK + k4;
        d[0] = f2bf(v.x); d[1] = f2bf(v.y); d[2] = f2bf(v.z); d[3] = f2bf(v.w);
    }
#if HAVE_ASYNC_LDS
#if __has_builtin(__builtin_amdgcn_s_wait_asynccnt)
    __builtin_amdgcn_s_wait_asynccnt(0);
#else
    asm volatile("s_wait_asynccnt 0x0" ::: "memory");
#endif
#endif
    __syncthreads();

    const int wave  = tid >> 5;
    const int lane  = tid & 31;
    const int k8    = (lane >> 4) << 3;    // +8 K-offset AND +8 M-offset (hi lanes)
    const int nlane = lane & 15;
    const int t0    = wave * 16;           // this wave's 16 output rows

    for (int ct = 0; ct < NOUT / 16; ++ct) {
        const int col0 = ct * 16;
        v8f acc = {};
#pragma unroll
        for (int ks = 0; ks < K / 32; ++ks) {
            int kb = ks * 32 + k8;
            v16bf a = load_frag<LDK>(lA, t0 + nlane, kb);
            v16bf b = load_frag<LDK>(lB, col0 + nlane, kb);
            acc = wmma_bf16(a, b, acc);
        }
#pragma unroll
        for (int i = 0; i < 8; ++i) {
            int  m   = i + k8;
            int  col = col0 + nlane;
            long tok = (long)tile * 64 + t0 + m;
            float v  = acc[i] + bias[bbase + col];
            if (DO_PE) {
                int   t   = (int)(tok & 127);
                float j   = (float)(col & ~1);
                float div = __expf(-9.210340371976184f * j / (float)NOUT);
                float ang = (float)t * div;
                v += (col & 1) ? __cosf(ang) : __sinf(ang);
            }
            if (DO_GELU) v = gelu_tanh(v);
            long orow = PER_NODE ? (tok * nodes + node) : tok;
            long oidx = orow * NOUT + col;
            if (DO_RES) v += res[oidx];
            if (OUT_BF) ((bf16*)outv)[oidx] = f2bf(v);
            else        ((float*)outv)[oidx] = v;
        }
    }
}

// ---------------------------------------------------------------------------
// LayerNorm over last dim (128).  One wave per row, 8 rows per block.
// ---------------------------------------------------------------------------
template<bool PER_NODE>
__global__ __launch_bounds__(256)
void layernorm_k(const float* __restrict__ X, const float* __restrict__ S,
                 const float* __restrict__ Bv, float* __restrict__ Y, int nodes)
{
    const int  lane = threadIdx.x & 31;
    const int  wrow = threadIdx.x >> 5;
    const long row  = (long)blockIdx.x * 8 + wrow;
    const float* x  = X + row * 128;

    float v[4]; float s = 0.f;
#pragma unroll
    for (int i = 0; i < 4; ++i) { v[i] = x[lane + i * 32]; s += v[i]; }
#pragma unroll
    for (int o = 16; o; o >>= 1) s += __shfl_xor(s, o, 32);
    float mu = s * (1.0f / 128.0f);
    float q = 0.f;
#pragma unroll
    for (int i = 0; i < 4; ++i) { float d = v[i] - mu; q += d * d; }
#pragma unroll
    for (int o = 16; o; o >>= 1) q += __shfl_xor(q, o, 32);
    float inv = rsqrtf(q * (1.0f / 128.0f) + 1e-6f);

    const long soff = PER_NODE ? (row % nodes) * 128 : 0;
#pragma unroll
    for (int i = 0; i < 4; ++i) {
        int col = lane + i * 32;
        Y[row * 128 + col] = (v[i] - mu) * inv * S[soff + col] + Bv[soff + col];
    }
}

// ---------------------------------------------------------------------------
// Transformer attention: one block per (b, n, h).  T=128, HD=32.
// qkv is bf16, row = ((b*T+t)*64+n)*384, cols [3][H=4][HD=32].
// ---------------------------------------------------------------------------
__global__ __launch_bounds__(256)
void attn_tr(const bf16* __restrict__ qkv, float* __restrict__ O)
{
    constexpr int T = 128, NN = 64;
    __shared__ __align__(16) bf16 lQ[T * 40];
    __shared__ __align__(16) bf16 lK[T * 40];
    __shared__ __align__(16) bf16 lVt[32 * 136];    // V transposed: [hd][t]
    __shared__ __align__(16) bf16 lP[T * 136];      // softmax(P) bf16
    __shared__ float lS[T][132];

    const int blk = blockIdx.x;
    const int b   = blk >> 8;
    const int n   = (blk >> 2) & 63;
    const int h   = blk & 3;
    const int tid = threadIdx.x;

    // Q/K: 16-byte chunk copies; V: element-wise transpose
    for (int idx = tid; idx < T * 4; idx += 256) {
        int t = idx >> 2, c = (idx & 3) * 8;
        long r = ((long)(b * T + t) * NN + n) * 384 + h * 32 + c;
        *(v8bf*)(lQ + t * 40 + c) = *(const v8bf*)(qkv + r);
        *(v8bf*)(lK + t * 40 + c) = *(const v8bf*)(qkv + r + 128);
    }
    for (int idx = tid; idx < T * 32; idx += 256) {
        int t = idx >> 5, hd = idx & 31;
        lVt[hd * 136 + t] = qkv[((long)(b * T + t) * NN + n) * 384 + 256 + h * 32 + hd];
    }
    __syncthreads();

    const int wave = tid >> 5, lane = tid & 31;
    const int k8 = (lane >> 4) << 3, nlane = lane & 15;
    const int t0 = wave * 16;
    const float scale = 0.17677669529663687f;       // 1/sqrt(32)

    // S = Q K^T * scale
    for (int ct = 0; ct < 8; ++ct) {
        v8f acc = {};
        v16bf a  = load_frag<40>(lQ, t0 + nlane, k8);
        v16bf bb = load_frag<40>(lK, ct * 16 + nlane, k8);
        acc = wmma_bf16(a, bb, acc);
#pragma unroll
        for (int i = 0; i < 8; ++i)
            lS[t0 + i + k8][ct * 16 + nlane] = acc[i] * scale;
    }
    __syncthreads();

    if (tid < T) {                                  // row softmax
        float mx = -1e30f;
        for (int m = 0; m < T; ++m) mx = fmaxf(mx, lS[tid][m]);
        float sum = 0.f;
        for (int m = 0; m < T; ++m) { float e = __expf(lS[tid][m] - mx); sum += e; lS[tid][m] = e; }
        float inv = 1.0f / sum;
        for (int m = 0; m < T; ++m) lP[tid * 136 + m] = f2bf(lS[tid][m] * inv);
    }
    __syncthreads();

    // O = P V   (16 rows x 32 cols per wave, K = 128)
    for (int ct = 0; ct < 2; ++ct) {
        v8f acc = {};
#pragma unroll
        for (int ks = 0; ks < 4; ++ks) {
            int kb = ks * 32 + k8;
            v16bf a  = load_frag<136>(lP, t0 + nlane, kb);
            v16bf bb = load_frag<136>(lVt, ct * 16 + nlane, kb);
            acc = wmma_bf16(a, bb, acc);
        }
#pragma unroll
        for (int i = 0; i < 8; ++i) {
            int t = t0 + i + k8;
            O[((long)(b * T + t) * NN + n) * 128 + h * 32 + ct * 16 + nlane] = acc[i];
        }
    }
}

// ---------------------------------------------------------------------------
// GRAND diffusion step: one block per (b,t).  N=64 nodes, D=128.
// qk bf16, row = (bt*64+n)*256, cols [2][H=4][HD=32].
// z_out = z + dt * (mean_h softmax(QK^T/sqrt(HD)) @ z - z)
// ---------------------------------------------------------------------------
__global__ __launch_bounds__(256)
void attn_grand(const float* __restrict__ Z, const bf16* __restrict__ QK,
                float* __restrict__ Zout, float dt)
{
    constexpr int NN = 64, D = 128;
    __shared__ __align__(16) bf16 lQ[NN * 40];
    __shared__ __align__(16) bf16 lK[NN * 40];
    __shared__ __align__(16) bf16 lZt[D * 72];      // z transposed: [d][m]
    __shared__ __align__(16) bf16 lAbf[NN * 72];
    __shared__ float lS[NN][68];
    __shared__ float lAcc[NN][68];

    const long bt  = blockIdx.x;
    const int  tid = threadIdx.x;
    const int  wave = tid >> 5, lane = tid & 31;
    const int  k8 = (lane >> 4) << 3, nlane = lane & 15;
    const float scale = 0.17677669529663687f;

    for (int idx = tid; idx < NN * 68; idx += 256) lAcc[idx / 68][idx % 68] = 0.f;
    for (int idx = tid; idx < NN * D; idx += 256) {
        int m = idx >> 7, d = idx & 127;
        lZt[d * 72 + m] = f2bf(Z[(bt * NN + m) * D + d]);
    }

    for (int h = 0; h < 4; ++h) {
        __syncthreads();
        for (int idx = tid; idx < NN * 4; idx += 256) {
            int m = idx >> 2, c = (idx & 3) * 8;
            long r = (bt * NN + m) * 256 + h * 32 + c;
            *(v8bf*)(lQ + m * 40 + c) = *(const v8bf*)(QK + r);
            *(v8bf*)(lK + m * 40 + c) = *(const v8bf*)(QK + r + 128);
        }
        __syncthreads();
        // 16 tiles of the 64x64 S over 8 waves (2 each)
#pragma unroll
        for (int rep = 0; rep < 2; ++rep) {
            int tix = wave + rep * 8;
            int rt = tix >> 2, ct = tix & 3;
            v8f acc = {};
            v16bf a  = load_frag<40>(lQ, rt * 16 + nlane, k8);
            v16bf bb = load_frag<40>(lK, ct * 16 + nlane, k8);
            acc = wmma_bf16(a, bb, acc);
#pragma unroll
            for (int i = 0; i < 8; ++i)
                lS[rt * 16 + i + k8][ct * 16 + nlane] = acc[i] * scale;
        }
        __syncthreads();
        if (tid < NN) {                             // softmax + head average
            float mx = -1e30f;
            for (int m = 0; m < NN; ++m) mx = fmaxf(mx, lS[tid][m]);
            float sum = 0.f;
            for (int m = 0; m < NN; ++m) { float e = __expf(lS[tid][m] - mx); sum += e; lS[tid][m] = e; }
            float inv = 0.25f / sum;
            for (int m = 0; m < NN; ++m) lAcc[tid][m] += lS[tid][m] * inv;
        }
    }
    __syncthreads();
    for (int idx = tid; idx < NN * NN; idx += 256)
        lAbf[(idx >> 6) * 72 + (idx & 63)] = f2bf(lAcc[idx >> 6][idx & 63]);
    __syncthreads();

    // z' = z + dt*(A@z - z): wave w -> column tile w (16 cols), rows 0..63
    const int ct = wave;
    for (int rt = 0; rt < 4; ++rt) {
        v8f acc = {};
#pragma unroll
        for (int ks = 0; ks < 2; ++ks) {
            int kb = ks * 32 + k8;
            v16bf a  = load_frag<72>(lAbf, rt * 16 + nlane, kb);
            v16bf bb = load_frag<72>(lZt, ct * 16 + nlane, kb);
            acc = wmma_bf16(a, bb, acc);
        }
#pragma unroll
        for (int i = 0; i < 8; ++i) {
            int  m = rt * 16 + i + k8;
            int  d = ct * 16 + nlane;
            long zi = (bt * NN + m) * D + d;
            float z0 = Z[zi];
            Zout[zi] = z0 + dt * (acc[i] - z0);
        }
    }
}

// ---------------------------------------------------------------------------
extern "C" void kernel_launch(void* const* d_in, const int* in_sizes, int n_in,
                              void* d_out, int out_size, void* d_ws, size_t ws_size,
                              hipStream_t stream)
{
    (void)in_sizes; (void)n_in; (void)out_size; (void)ws_size;
    const float* x      = (const float*)d_in[0];
    const float* tWi    = (const float*)d_in[1];
    const float* tbi    = (const float*)d_in[2];
    const float* tWqkv  = (const float*)d_in[3];
    const float* tbqkv  = (const float*)d_in[4];
    const float* tWo    = (const float*)d_in[5];
    const float* tbo    = (const float*)d_in[6];
    const float* ln1s   = (const float*)d_in[7];
    const float* ln1b   = (const float*)d_in[8];
    const float* ln2s   = (const float*)d_in[9];
    const float* ln2b   = (const float*)d_in[10];
    const float* tW1    = (const float*)d_in[11];
    const float* tb1    = (const float*)d_in[12];
    const float* tW2    = (const float*)d_in[13];
    const float* tb2    = (const float*)d_in[14];
    const float* gWin   = (const float*)d_in[15];
    const float* gbin   = (const float*)d_in[16];
    const float* gWqk   = (const float*)d_in[17];
    const float* gbqk   = (const float*)d_in[18];
    const float* gWout  = (const float*)d_in[19];
    const float* gbout  = (const float*)d_in[20];
    const float* lns    = (const float*)d_in[21];
    const float* lnb    = (const float*)d_in[22];
    const float* mW1    = (const float*)d_in[23];
    const float* mb1    = (const float*)d_in[24];
    const float* mW2    = (const float*)d_in[25];
    const float* mb2    = (const float*)d_in[26];
    float* out = (float*)d_out;

    const long TOK = 65536;                                // B*T*N
    const size_t FB = (size_t)TOK * 128 * sizeof(float);   // 32 MB
    char*  ws   = (char*)d_ws;
    float* bx0  = (float*)(ws);
    float* bh   = (float*)(ws + FB);
    float* bo   = (float*)(ws + 2 * FB);
    float* bx1  = (float*)(ws + 3 * FB);
    bf16*  bqkv = (bf16*)(ws + 4 * FB);                    // 48 MB bf16, reused for qk

    // pre-transposed bf16 weights (prep_w layout), ~15 MB
    size_t off = 4 * FB + (size_t)TOK * 384 * 2;
    bf16* wtWi   = (bf16*)(ws + off); off += (size_t)64 * 128 * 40  * 2;
    bf16* wtWqkv = (bf16*)(ws + off); off += (size_t)64 * 384 * 136 * 2;
    bf16* wtWo   = (bf16*)(ws + off); off += (size_t)64 * 128 * 136 * 2;
    bf16* wtW1   = (bf16*)(ws + off); off += (size_t)64 * 128 * 136 * 2;
    bf16* wtW2   = (bf16*)(ws + off); off += (size_t)64 * 128 * 136 * 2;
    bf16* wtGin  = (bf16*)(ws + off); off += (size_t)128 * 136 * 2;
    bf16* wtGqk  = (bf16*)(ws + off); off += (size_t)256 * 136 * 2;
    bf16* wtGout = (bf16*)(ws + off); off += (size_t)128 * 136 * 2;
    bf16* wtM1   = (bf16*)(ws + off); off += (size_t)128 * 136 * 2;
    bf16* wtM2   = (bf16*)(ws + off);

    auto nblk = [](long total) { return dim3((unsigned)((total + 255) / 256)); };
    prep_w<32, 128> <<<nblk((long)64 * 128 * 40),  256, 0, stream>>>(tWi,   wtWi,   64);
    prep_w<128, 384><<<nblk((long)64 * 384 * 136), 256, 0, stream>>>(tWqkv, wtWqkv, 64);
    prep_w<128, 128><<<nblk((long)64 * 128 * 136), 256, 0, stream>>>(tWo,   wtWo,   64);
    prep_w<128, 128><<<nblk((long)64 * 128 * 136), 256, 0, stream>>>(tW1,   wtW1,   64);
    prep_w<128, 128><<<nblk((long)64 * 128 * 136), 256, 0, stream>>>(tW2,   wtW2,   64);
    prep_w<128, 128><<<nblk((long)128 * 136),      256, 0, stream>>>(gWin,  wtGin,  1);
    prep_w<128, 256><<<nblk((long)256 * 136),      256, 0, stream>>>(gWqk,  wtGqk,  1);
    prep_w<128, 128><<<nblk((long)128 * 136),      256, 0, stream>>>(gWout, wtGout, 1);
    prep_w<128, 128><<<nblk((long)128 * 136),      256, 0, stream>>>(mW1,   wtM1,   1);
    prep_w<128, 128><<<nblk((long)128 * 136),      256, 0, stream>>>(mW2,   wtM2,   1);

    const dim3 blk(128);
    const dim3 gPN(16, 64);                         // per-node: 1024/64 tiles x 64 nodes
    const dim3 gSH(1024, 1);                        // shared: 65536/64 tiles

    // ---- per-node transformer (vmapped over N) ----
    gemm_ep<32, 128, true, false, false, true, false>
        <<<gPN, blk, 0, stream>>>(x, wtWi, tbi, nullptr, bx0, 64);
    layernorm_k<true><<<8192, 256, 0, stream>>>(bx0, ln1s, ln1b, bh, 64);
    gemm_ep<128, 384, true, false, false, false, true>
        <<<gPN, blk, 0, stream>>>(bh, wtWqkv, tbqkv, nullptr, bqkv, 64);
    attn_tr<<<2048, 256, 0, stream>>>(bqkv, bo);
    gemm_ep<128, 128, true, false, true, false, false>
        <<<gPN, blk, 0, stream>>>(bo, wtWo, tbo, bx0, bx1, 64);
    layernorm_k<true><<<8192, 256, 0, stream>>>(bx1, ln2s, ln2b, bh, 64);
    gemm_ep<128, 128, true, true, false, false, false>
        <<<gPN, blk, 0, stream>>>(bh, wtW1, tb1, nullptr, bo, 64);
    gemm_ep<128, 128, true, false, true, false, false>
        <<<gPN, blk, 0, stream>>>(bo, wtW2, tb2, bx1, bx0, 64);

    // ---- GRAND graph diffusion ----
    gemm_ep<128, 128, false, false, false, false, false>
        <<<gSH, blk, 0, stream>>>(bx0, wtGin, gbin, nullptr, bh, 1);
    float* zc = bh; float* zn = bx1;
    const float dt = 1.0f / 6.0f;                   // linspace(0,0.5,4) deltas
    for (int i = 0; i < 3; ++i) {
        gemm_ep<128, 256, false, false, false, false, true>
            <<<gSH, blk, 0, stream>>>(zc, wtGqk, gbqk, nullptr, bqkv, 1);
        attn_grand<<<1024, 256, 0, stream>>>(zc, bqkv, zn, dt);
        float* t = zc; zc = zn; zn = t;
    }
    // zc == bx1 holds final z
    gemm_ep<128, 128, false, false, false, false, false>
        <<<gSH, blk, 0, stream>>>(zc, wtGout, gbout, nullptr, bo, 1);

    // ---- head MLP ----
    layernorm_k<false><<<8192, 256, 0, stream>>>(bo, lns, lnb, bx0, 1);
    gemm_ep<128, 128, false, true, false, false, false>
        <<<gSH, blk, 0, stream>>>(bx0, wtM1, mb1, nullptr, bh, 1);
    gemm_ep<128, 128, false, false, false, false, false>
        <<<gSH, blk, 0, stream>>>(bh, wtM2, mb2, nullptr, out, 1);
}
